// ValueVectorInputLayer_29240137351865
// MI455X (gfx1250) — matrix operand order repaired and checked by reading, not compiled, not run on hardware
//
#include <hip/hip_runtime.h>
#include <hip/hip_bf16.h>

// ValueVectorInputLayer on MI455X (gfx1250):
//   stage 1: pack sign(values) -> f16 table [16 x 2048] (rows 10..15 zero pad)
//   stage 2: 25 tiny GEMMs via v_wmma_f32_16x16x32_f16, batched 5 taps/wave:
//            G[kk][n][o] = sum_i sign(values[n,i]) * sign(w[o,i,kk])
//   stage 3: out[b,o,y,x] = sign(bias[o] + sum_kk G[kk][idx(b,y,x,kk)][o])
// Roofline: factoring the 10-entry embedding out of the conv collapses
// 483 GFLOP -> ~4 GFLOP of WMMA + 0.12 GFLOP LUT-sum; the kernel is bound by
// reading w once (419 MB @ 23.3 TB/s ~ 18 us). Sign-bit tricks (not compares)
// keep the f32->(+/-1)f16 conversion at ~2 integer VALU ops/element so VALU
// does not dominate the HBM stream.

typedef __attribute__((ext_vector_type(16))) _Float16 v16h;
typedef __attribute__((ext_vector_type(8)))  _Float16 v8h_t;
typedef __attribute__((ext_vector_type(8)))  float    v8f;
typedef unsigned int u4_u __attribute__((ext_vector_type(4), aligned(4)));  // 4B-aligned u32x4

#define D_CH   2048
#define N_VAL  10
#define KK     25      // 5x5 taps
#define BATCH  16
#define H_IN   28
#define H_OUT  12

__device__ __forceinline__ _Float16 sign_h_exact(float f) {
    return (f > 0.0f) ? (_Float16)1.0f : ((f < 0.0f) ? (_Float16)(-1.0f) : (_Float16)0.0f);
}

// ---------------- stage 1: binarize value vectors to f16 -------------------
__global__ void vv_pack_values(const float* __restrict__ values,
                               _Float16* __restrict__ sV) {
    int t = blockIdx.x * blockDim.x + threadIdx.x;     // over 16*2048 (padded rows)
    if (t < 16 * D_CH) {
        int n = t / D_CH;
        float v = (n < N_VAL) ? values[t] : 0.0f;
        sV[t] = sign_h_exact(v);                        // exact: sign(0)=0 like jnp.sign
    }
}

// ---------------- stage 2: GEMMs, M=16(10 valid) N=2048 K=2048, 5 taps/wave
// grid = (5 ky, 16), block = 256 (8 waves). Wave: one 16x16 tile x 5 kx taps.
__global__ void vv_gemm_sign(const float* __restrict__ w,        // [2048][2048][25] floats
                             const _Float16* __restrict__ sV,    // [16][2048] f16 (+/-1)
                             float* __restrict__ G) {            // [25][16][2048] f32
    const int ky    = blockIdx.x;                 // 0..4 kernel row (5 kx taps each)
    const int wave  = threadIdx.x >> 5;           // 0..7
    const int lane  = threadIdx.x & 31;
    const int grp   = lane >> 4;                  // 0: lanes 0-15, 1: lanes 16-31
    const int row   = lane & 15;                  // A row (n) / B col (o_local)
    const int otile = blockIdx.y * 8 + wave;      // 0..127
    const int o     = otile * 16 + row;           // output channel column

    v8f c[5] = {};                                // 5 independent f32 accum tiles

    const _Float16* aRow  = sV + (size_t)row * D_CH;
    // w[o][k][ky*5 + kx]: k-stride = 25 floats; 5 kx taps are contiguous.
    const float*    wBase = w + (size_t)o * D_CH * KK + ky * 5;

    for (int k0 = 0; k0 < D_CH; k0 += 32) {
        // ---- A fragment (16x32 f16), contiguous K layout (matches B):
        //   lanes 0-15:  K = k0+{0..15};  lanes 16-31: K = k0+{16..31}
        v8h_t alo = *(const v8h_t*)(aRow + k0 + grp * 16);
        v8h_t ahi = *(const v8h_t*)(aRow + k0 + grp * 16 + 8);
        v16h a;
#pragma unroll
        for (int j = 0; j < 8; ++j) { a[j] = alo[j]; a[8 + j] = ahi[j]; }

        // ---- 5 B fragments (32x16 f16): N = row, K = k0 + grp*16 + j.
        // Per (j, lane): one B128 + one B32 covers all 5 taps (20 contiguous bytes).
        // sign(f) as f16 via sign bit: 0x3C00 | (bits>>16 & 0x8000); two elems/dword.
        union { v16h h; unsigned int u[8]; } bf[5];
        const float* wp = wBase + (size_t)(k0 + grp * 16) * KK;
#pragma unroll
        for (int d = 0; d < 8; ++d) {
            const unsigned int* p0 = (const unsigned int*)(wp + (size_t)(2 * d)     * KK);
            const unsigned int* p1 = (const unsigned int*)(wp + (size_t)(2 * d + 1) * KK);
            u4_u lo0 = *(const u4_u*)p0;  unsigned int e0 = p0[4];
            u4_u lo1 = *(const u4_u*)p1;  unsigned int e1 = p1[4];
            unsigned int s0[5] = { lo0[0], lo0[1], lo0[2], lo0[3], e0 };
            unsigned int s1[5] = { lo1[0], lo1[1], lo1[2], lo1[3], e1 };
#pragma unroll
            for (int t = 0; t < 5; ++t)
                bf[t].u[d] = 0x3C003C00u
                           | ((s0[t] >> 16) & 0x00008000u)    // low  half: K even
                           | ( s1[t]        & 0x80000000u);   // high half: K odd
        }

        // prefetch next K-slab of w (global_prefetch_b8)
        __builtin_prefetch(wp + 32 * KK, 0, 1);

        // 5 independent D = A x B + C  -> 5x v_wmma_f32_16x16x32_f16 per iter
#pragma unroll
        for (int t = 0; t < 5; ++t)
            c[t] = __builtin_amdgcn_wmma_f32_16x16x32_f16(false, a, false, bf[t].h,
                                                          (short)0, c[t], false, false);
    }

    // ---- store: lane covers rows M = v + 8*grp, col N = row, taps kk = ky*5+t
#pragma unroll
    for (int t = 0; t < 5; ++t) {
        float* Gp = G + (size_t)(ky * 5 + t) * 16 * D_CH;
#pragma unroll
        for (int v = 0; v < 8; ++v)
            Gp[(size_t)(v + grp * 8) * D_CH + o] = c[t][v];
    }
}

// ---------------- stage 3: LUT gather-sum + bias + sign --------------------
// grid = (16*12*12, 2048/256), block = 256. One block = one pixel x 256 channels.
__global__ void vv_conv_lut(const float* __restrict__ x,      // [16][1][28][28]
                            const float* __restrict__ G,      // [25][16][2048]
                            const float* __restrict__ bias,   // [2048]
                            float* __restrict__ out) {        // [16][2048][12][12]
    __shared__ int sidx[KK];

    const int bp = blockIdx.x;                   // b*144 + oy*12 + ox
    const int b  = bp / (H_OUT * H_OUT);
    const int yx = bp % (H_OUT * H_OUT);
    const int oy = yx / H_OUT;
    const int ox = yx % H_OUT;

    if (threadIdx.x < KK) {
        const int ky = threadIdx.x / 5;
        const int kx = threadIdx.x % 5;
        const int iy = 2 * oy + ky;              // stride 2, VALID
        const int ix = 2 * ox + kx;
        const float v = x[((size_t)b * H_IN + iy) * H_IN + ix];
        sidx[threadIdx.x] = (int)(v * (float)(N_VAL - 1));   // trunc == .astype(int32), v>=0
    }
    __syncthreads();

    const int o = blockIdx.y * blockDim.x + threadIdx.x;
    float acc = bias[o];
#pragma unroll
    for (int k = 0; k < KK; ++k)
        acc += G[((size_t)k * 16 + sidx[k]) * D_CH + o];     // coalesced in o; G is L2-resident

    const float s = (acc > 0.0f) ? 1.0f : ((acc < 0.0f) ? -1.0f : 0.0f);
    out[((size_t)(b * D_CH + o) * H_OUT + oy) * H_OUT + ox] = s;
}

extern "C" void kernel_launch(void* const* d_in, const int* in_sizes, int n_in,
                              void* d_out, int out_size, void* d_ws, size_t ws_size,
                              hipStream_t stream) {
    const float* x      = (const float*)d_in[0];   // [16,1,28,28]
    const float* values = (const float*)d_in[1];   // [10,2048]
    const float* w      = (const float*)d_in[2];   // [2048,2048,5,5]
    const float* bias   = (const float*)d_in[3];   // [2048]
    float* out = (float*)d_out;                    // [16,2048,12,12] f32

    // workspace: sV (64 KB f16) | G (3.2 MB f32)
    _Float16* sV = (_Float16*)d_ws;
    float*    G  = (float*)((char*)d_ws + (size_t)16 * D_CH * sizeof(_Float16));

    vv_pack_values<<<dim3((16 * D_CH + 255) / 256), dim3(256), 0, stream>>>(values, sV);
    vv_gemm_sign<<<dim3(5, 16), dim3(256), 0, stream>>>(w, sV, G);
    vv_conv_lut<<<dim3(BATCH * H_OUT * H_OUT, D_CH / 256), dim3(256), 0, stream>>>(x, G, bias, out);
}
